// TreesLayer_59837484367925
// MI455X (gfx1250) — matrix operand order repaired and checked by reading, not compiled, not run on hardware
//
#include <hip/hip_runtime.h>

typedef __attribute__((ext_vector_type(16))) _Float16 v16h;
typedef __attribute__((ext_vector_type(8)))  _Float16 v8h;
typedef __attribute__((ext_vector_type(8)))  float    v8f;

#define TREES   16
#define IN_DIM  2048
#define OUT_DIM 512
#define NLEAF   16
#define NCOL    256   // padded (tree, node) columns: 16 trees * 16 (15 nodes + 1 zero pad)
#define BATCH   8192

__device__ __forceinline__ void f32_split(float w, _Float16 &h, _Float16 &l) {
  h = (_Float16)w;
  l = (_Float16)(w - (float)h);   // residual is exactly representable; h+l ~ f32-accurate
}

// node_weights [T][I][15] f32 -> wnh/wnl [col=256][K=2048] f16 (hi/lo split),
// col = t*16 + n, n==15 column is zero padding.
__global__ void tel_conv_node(const float* __restrict__ nw,
                              _Float16* __restrict__ wnh,
                              _Float16* __restrict__ wnl) {
  int idx = blockIdx.x * blockDim.x + threadIdx.x;   // NCOL * IN_DIM
  if (idx >= NCOL * IN_DIM) return;
  int col = idx >> 11;
  int k   = idx & (IN_DIM - 1);
  int t = col >> 4, n = col & 15;
  float w = (n < 15) ? nw[((size_t)t * IN_DIM + k) * 15 + n] : 0.0f;
  _Float16 h, l;
  f32_split(w, h, l);
  wnh[idx] = h;
  wnl[idx] = l;
}

// leaf_weights [T][O][16] f32 -> wb [o=512][k=256] f16 with k = t*16 + l
__global__ void tel_conv_leaf(const float* __restrict__ lw,
                              _Float16* __restrict__ wb) {
  int idx = blockIdx.x * blockDim.x + threadIdx.x;   // OUT_DIM * NCOL
  if (idx >= OUT_DIM * NCOL) return;
  int o = idx >> 8;
  int k = idx & 255;
  int t = k >> 4, l = k & 15;
  wb[idx] = (_Float16)lw[((size_t)t * OUT_DIM + o) * NLEAF + l];
}

__global__ void __launch_bounds__(128) tel_main(const float* __restrict__ x,
                                                const _Float16* __restrict__ wnh,
                                                const _Float16* __restrict__ wnl,
                                                const _Float16* __restrict__ wb,
                                                float* __restrict__ out) {
  __shared__ float s_lds[16 * 256];                     // [row][tree*16 + node], f32 gates
  __shared__ __align__(32) _Float16 p_lds[16 * 256];    // [row][t*16 + l], f16 leaf probs

  const int tid  = threadIdx.x;
  const int lane = tid & 31;
  const int wv   = tid >> 5;        // wave 0..3
  const int half = lane >> 4;       // 0 | 1 (lane half)
  const int lc   = lane & 15;
  const int m0   = blockIdx.x * 16; // batch row tile

  // ---------------- Stage 1: a = X @ Wnode via split-f16 WMMA ----------------
  // This wave handles trees tbase..tbase+3 (N tile = 16 padded node columns).
  v8f acc[4] = {};
  const int tbase = wv * 4;
  const float* xrow = x + (size_t)(m0 + lc) * IN_DIM;

  for (int kt = 0; kt < IN_DIM / 32; ++kt) {
    const int k0 = kt * 32;
    const int c0 = k0 + half * 16;  // A layout: half 0 -> K 0..15; half 1 -> K 16..31 (contiguous)
    float4 f0 = *(const float4*)(xrow + c0);
    float4 f1 = *(const float4*)(xrow + c0 + 4);
    float4 f2 = *(const float4*)(xrow + c0 + 8);
    float4 f3 = *(const float4*)(xrow + c0 + 12);
    __builtin_prefetch(xrow + c0 + 64, 0, 0);   // global_prefetch_b8, 2 k-tiles ahead

    float xs[16] = { f0.x, f0.y, f0.z, f0.w, f1.x, f1.y, f1.z, f1.w,
                     f2.x, f2.y, f2.z, f2.w, f3.x, f3.y, f3.z, f3.w };
    v16h ah, al;
#pragma unroll
    for (int i = 0; i < 16; ++i) {
      _Float16 h, l;
      f32_split(xs[i], h, l);
      ah[i] = h;
      al[i] = l;
    }

#pragma unroll
    for (int tt = 0; tt < 4; ++tt) {
      // B layout: lanes = N cols; half 0 holds K 0..15, half 1 holds K 16..31 (contiguous)
      const int off = ((tbase + tt) * 16 + lc) * IN_DIM + k0 + half * 16;
      v16h bh = *(const v16h*)(wnh + off);
      v16h bl = *(const v16h*)(wnl + off);
      acc[tt] = __builtin_amdgcn_wmma_f32_16x16x32_f16(false, ah, false, bh, (short)0, acc[tt], false, false);
      acc[tt] = __builtin_amdgcn_wmma_f32_16x16x32_f16(false, al, false, bh, (short)0, acc[tt], false, false);
      acc[tt] = __builtin_amdgcn_wmma_f32_16x16x32_f16(false, ah, false, bl, (short)0, acc[tt], false, false);
    }
  }

  // smooth-step gates -> LDS. C layout: lane holds node lc; VGPR r = row r + 8*half.
#pragma unroll
  for (int tt = 0; tt < 4; ++tt) {
    v8f a = acc[tt];
#pragma unroll
    for (int r = 0; r < 8; ++r) {
      float t01 = fminf(fmaxf(a[r] + 0.5f, 0.0f), 1.0f);   // SMOOTH_STEP_PARAM = 1.0
      float s = t01 * t01 * (3.0f - 2.0f * t01);
      int row = r + half * 8;
      s_lds[row * 256 + (tbase + tt) * 16 + lc] = s;
    }
  }
  __syncthreads();

  // leaf probabilities: product of 4 gates along heap path, written in f16 A-fragment order
  for (int p = tid; p < 16 * 256; p += 128) {   // uniform trip count: no divergence
    int row = p >> 8;
    int k = p & 255;
    int t = k >> 4, l = k & 15;
    const float* sb = &s_lds[row * 256 + t * 16];
    float prob = 1.0f;
#pragma unroll
    for (int lvl = 0; lvl < 4; ++lvl) {
      int n   = (1 << lvl) - 1 + (l >> (4 - lvl));
      int bit = (l >> (3 - lvl)) & 1;
      float s = sb[n];
      prob *= bit ? (1.0f - s) : s;
    }
    p_lds[p] = (_Float16)prob;
  }
  __syncthreads();

  // ---------------- Stage 2: out = P @ Wleaf, K = 256 (all trees x leaves) ----------------
  // Preload all 8 K-tile A-fragments from LDS (row = lc for both lane halves).
  v16h afr[8];
#pragma unroll
  for (int kt = 0; kt < 8; ++kt) {
    const _Float16* pp = p_lds + lc * 256 + kt * 32 + half * 16;
    v8h lo8 = *(const v8h*)(pp);        // K kt*32+{0..7 | 16..23}
    v8h hi8 = *(const v8h*)(pp + 8);    // K kt*32+{8..15 | 24..31}
#pragma unroll
    for (int i = 0; i < 8; ++i) {
      afr[kt][i]     = lo8[i];
      afr[kt][i + 8] = hi8[i];
    }
  }

  for (int j = 0; j < 8; ++j) {
    const int ob = (wv * 8 + j) * 16;   // waves cover all 32 output tiles
    v8f oacc = {};
#pragma unroll
    for (int kt = 0; kt < 8; ++kt) {
      const _Float16* wp = wb + (size_t)(ob + lc) * 256 + kt * 32 + half * 16;
      v16h bf = *(const v16h*)wp;
      oacc = __builtin_amdgcn_wmma_f32_16x16x32_f16(false, afr[kt], false, bf, (short)0, oacc, false, false);
    }
#pragma unroll
    for (int r = 0; r < 8; ++r) {
      out[(size_t)(m0 + r + half * 8) * OUT_DIM + ob + lc] = oacc[r];
    }
  }
}

extern "C" void kernel_launch(void* const* d_in, const int* in_sizes, int n_in,
                              void* d_out, int out_size, void* d_ws, size_t ws_size,
                              hipStream_t stream) {
  (void)in_sizes; (void)n_in; (void)out_size; (void)ws_size;
  const float* x  = (const float*)d_in[0];
  const float* nw = (const float*)d_in[1];   // [16][2048][15]
  const float* lw = (const float*)d_in[2];   // [16][512][16]
  float* out = (float*)d_out;

  // workspace: 1 MB hi + 1 MB lo node weights + 256 KB leaf weights (f16)
  _Float16* wnh = (_Float16*)d_ws;
  _Float16* wnl = wnh + (size_t)NCOL * IN_DIM;
  _Float16* wb  = wnl + (size_t)NCOL * IN_DIM;

  tel_conv_node<<<(NCOL * IN_DIM + 255) / 256, 256, 0, stream>>>(nw, wnh, wnl);
  tel_conv_leaf<<<(OUT_DIM * NCOL + 255) / 256, 256, 0, stream>>>(lw, wb);
  tel_main<<<BATCH / 16, 128, 0, stream>>>(x, wnh, wnl, wb, out);
}
